// GaussianMixtureAccumulator_81346680586527
// MI455X (gfx1250) — matrix operand order repaired and checked by reading, not compiled, not run on hardware
//
#include <hip/hip_runtime.h>
#include <hip/hip_bf16.h>

typedef __attribute__((ext_vector_type(16))) _Float16 v16h;
typedef __attribute__((ext_vector_type(2)))  _Float16 h2;
typedef __attribute__((ext_vector_type(8)))  float    v8f;

#define K_NB   32          // neighbors per query (== wave32 lanes, == WMMA K)
#define C_FT   64          // feature channels
#define NCH    80          // 64 feat + 4 semantic + 12 zero pad -> 5 blocks of 16
#define CH_STR 34          // halfs per channel row: 32 k + 2 pad = 68 B (17 dwords, coprime 64 banks)
#define WPB    8           // waves (queries) per 256-thread block
#define TILE_H (NCH * CH_STR + 32)   // + 32-half weight row

__global__ __launch_bounds__(256)
void gma_wmma_kernel(const float* __restrict__ mu,
                     const float* __restrict__ scale,
                     const float* __restrict__ rot,
                     const float* __restrict__ feat,
                     const float* __restrict__ sem,
                     const float* __restrict__ qp,
                     const int*   __restrict__ nbi,
                     const unsigned char* __restrict__ nbm,
                     float* __restrict__ out,
                     int M)
{
    __shared__ _Float16 lds[WPB * TILE_H];

    const int lane = threadIdx.x & 31;
    const int wid  = threadIdx.x >> 5;
    const int m    = blockIdx.x * WPB + wid;
    if (m >= M) return;                 // whole wave exits together -> EXEC full below

    _Float16* tile = &lds[wid * TILE_H];
    _Float16* wrow = tile + NCH * CH_STR;

    // ---------------- Phase A: per-lane (lane == neighbor k) weight ----------------
    const int  k_flat = m * K_NB + lane;
    const int  idx    = nbi[k_flat];
    const bool valid  = nbm[k_flat] != 0;
    const int  safe   = valid ? idx : 0;

    const float* muP = mu    + (size_t)safe * 3;
    const float* scP = scale + (size_t)safe * 3;
    const float4 q   = ((const float4*)rot)[safe];   // (w,x,y,z)
    const float4 sm  = ((const float4*)sem)[safe];

    const float qx = qp[m * 3 + 0];                  // uniform per wave -> scalar loads
    const float qy = qp[m * 3 + 1];
    const float qz = qp[m * 3 + 2];

    float qn  = sqrtf(q.x*q.x + q.y*q.y + q.z*q.z + q.w*q.w);
    float inv = 1.0f / fmaxf(qn, 1e-12f);
    float w0 = q.x*inv, x = q.y*inv, y = q.z*inv, z = q.w*inv;

    float r00 = 1.f - 2.f*(y*y + z*z), r01 = 2.f*(x*y - z*w0), r02 = 2.f*(x*z + y*w0);
    float r10 = 2.f*(x*y + z*w0), r11 = 1.f - 2.f*(x*x + z*z), r12 = 2.f*(y*z - x*w0);
    float r20 = 2.f*(x*z - y*w0), r21 = 2.f*(y*z + x*w0), r22 = 1.f - 2.f*(x*x + y*y);

    float dx = qx - muP[0], dy = qy - muP[1], dz = qz - muP[2];
    float dl0 = r00*dx + r01*dy + r02*dz;
    float dl1 = r10*dx + r11*dy + r12*dz;
    float dl2 = r20*dx + r21*dy + r22*dz;

    float s0 = scP[0], s1 = scP[1], s2 = scP[2];
    float d2 = dl0*dl0 / (s0*s0 + 1e-8f)
             + dl1*dl1 / (s1*s1 + 1e-8f)
             + dl2*dl2 / (s2*s2 + 1e-8f);

    float wgt = __expf(-0.5f / (1.0f + 1e-8f) * d2) * (valid ? 1.0f : 0.0f);

    float s = wgt;                                   // wave32 reduction
    #pragma unroll
    for (int o = 16; o > 0; o >>= 1) s += __shfl_xor(s, o, 32);
    const float wn = wgt / (s + 1e-8f);

    // w output lives at offset 68*M (feats 64M, semantic 4M, w 32M)
    out[(size_t)M * 68 + (size_t)m * K_NB + lane] = wn;
    wrow[lane] = (_Float16)wn;

    // ---------------- Stage gathered tile in LDS, k-major (transposed) -------------
    const float4* frow = (const float4*)(feat + (size_t)safe * C_FT);
    #pragma unroll
    for (int t = 0; t < 16; ++t) {
        float4 f = frow[t];
        int c = 4 * t;
        tile[(c + 0) * CH_STR + lane] = (_Float16)f.x;
        tile[(c + 1) * CH_STR + lane] = (_Float16)f.y;
        tile[(c + 2) * CH_STR + lane] = (_Float16)f.z;
        tile[(c + 3) * CH_STR + lane] = (_Float16)f.w;
    }
    tile[64 * CH_STR + lane] = (_Float16)sm.x;       // semantic as channels 64..67
    tile[65 * CH_STR + lane] = (_Float16)sm.y;
    tile[66 * CH_STR + lane] = (_Float16)sm.z;
    tile[67 * CH_STR + lane] = (_Float16)sm.w;
    #pragma unroll
    for (int c = 68; c < NCH; ++c)                   // zero the pad channels
        tile[c * CH_STR + lane] = (_Float16)0.0f;

    // all LDS traffic is wave-private; just drain DS before re-reading
    asm volatile("s_wait_dscnt 0" ::: "memory");

    // ---------------- Phase B: 5x v_wmma_f32_16x16x32_f16 per query -----------------
    const int h = lane >> 4;      // lane half
    const int j = lane & 15;      // output column

    // A[i,k] = wn[k], rows replicated. A k order used here (16-bit, 16x32, contiguous, not the ISA table order):
    // half h: VGPR v -> K = 2v+16h, 2v+1+16h
    v16h A;
    #pragma unroll
    for (int v = 0; v < 8; ++v) {
        int k0 = 16 * h + 2 * v;
        h2 p = *(const h2*)(wrow + k0);              // broadcast per half (conflict-free)
        A[2 * v] = p.x; A[2 * v + 1] = p.y;
    }

    // B[k,j] = tile_f16[c0+j][k]. B k order used here (32x16, contiguous, same as A): half h holds K = 16h..16h+15,
    // packed 2/VGPR -> VGPR v holds K = 2v+16h, 2v+1+16h at column j = lane%16.
    v8f acc[5];
    #pragma unroll
    for (int cb = 0; cb < 5; ++cb) {
        v16h B;
        const _Float16* col = tile + (cb * 16 + j) * CH_STR + 16 * h;
        #pragma unroll
        for (int v = 0; v < 8; ++v) {
            h2 p = *(const h2*)(col + 2 * v);
            B[2 * v] = p.x; B[2 * v + 1] = p.y;
        }
        v8f c = {};
        acc[cb] = __builtin_amdgcn_wmma_f32_16x16x32_f16(
            /*neg_a=*/false, A, /*neg_b=*/false, B,
            /*c_mod=*/(short)0, c, /*reuse_a=*/false, /*reuse_b=*/false);
    }

    // D columns are the reduced outputs: every D VGPR/row holds out[c0 + j].
    float v01 = (lane < 16) ? acc[0][0] : acc[1][0]; // channels 0..31, fully coalesced
    out[(size_t)m * C_FT + lane] = v01;
    float v23 = (lane < 16) ? acc[2][0] : acc[3][0]; // channels 32..63
    out[(size_t)m * C_FT + 32 + lane] = v23;
    if (lane < 4)                                     // semantic (block 4, cols 0..3)
        out[(size_t)M * 64 + (size_t)m * 4 + lane] = acc[4][0];
}

extern "C" void kernel_launch(void* const* d_in, const int* in_sizes, int n_in,
                              void* d_out, int out_size, void* d_ws, size_t ws_size,
                              hipStream_t stream) {
    const float* mu   = (const float*)d_in[0];
    const float* scl  = (const float*)d_in[1];
    const float* rot  = (const float*)d_in[2];
    const float* feat = (const float*)d_in[3];
    const float* sem  = (const float*)d_in[4];
    const float* qp   = (const float*)d_in[5];
    // d_in[6] = voxel_coords: unused interface placeholder
    const int*   nbi  = (const int*)d_in[7];
    const unsigned char* nbm = (const unsigned char*)d_in[8];  // bool mask
    float* out = (float*)d_out;

    const int M = in_sizes[5] / 3;        // query count
    dim3 block(256);
    dim3 grid((M + WPB - 1) / WPB);
    gma_wmma_kernel<<<grid, block, 0, stream>>>(mu, scl, rot, feat, sem, qp,
                                                nbi, nbm, out, M);
}
